// SiMBAMusic_90254442758311
// MI455X (gfx1250) — compile-verified, matrix operand order checked by reading, not run on hardware
//
#include <hip/hip_runtime.h>
#include <math.h>

// ---------------------------------------------------------------------------
// Model dimensions (match reference)
// ---------------------------------------------------------------------------
#define LSEQ    2048
#define DMODEL  512
#define NLAYERS 8
#define DSTATE  16
#define DCONV   4
#define DINNER  1024
#define DXZ     2048            // 2*DINNER
#define DTRANK  32
#define NXDBL   64              // DT_RANK + 2*D_STATE
#define VOCAB   4096

typedef __attribute__((ext_vector_type(16))) __bf16 v16bf;
typedef __attribute__((ext_vector_type(8)))  float  v8f;

// round-to-nearest-even f32 -> bf16 bits
static __device__ __forceinline__ unsigned short f32_to_bf16(float f) {
    unsigned int x = __float_as_uint(f);
    unsigned int r = x + 0x7FFFu + ((x >> 16) & 1u);
    return (unsigned short)(r >> 16);
}

static __device__ __forceinline__ float sigmoidf_(float x) {
    return 1.0f / (1.0f + __expf(-x));
}

union Frag16 { v16bf v; uint4 q[2]; };

// Two 16-byte loads per fragment: K-groups kb+8*half and kb+16+8*half
// (ISA 7.12.2, 16-bit A 16x32 layout). ptr is pre-offset by half*8.
static __device__ __forceinline__ Frag16 load_frag(const unsigned short* __restrict__ p) {
    Frag16 f;
    f.q[0] = *(const uint4*)(p);
    f.q[1] = *(const uint4*)(p + 16);
    return f;
}

// ---------------------------------------------------------------------------
// WMMA GEMM: C[M,N] = A[M,K] * W[N,K]^T (+ R[M,N] if HASR)
// All dims/strides compile-time -> B-subtiles and the 2nd A-row are
// immediate offsets from two incrementing base pointers (no address spills).
// Each wave: 32x64 strip = 8 accumulators; per 32-wide K step: 12x
// global_load_b128 -> 8 back-to-back v_wmma_f32_16x16x32_bf16.
// ---------------------------------------------------------------------------
template <bool HASR, int MM, int NN, int KK, int LDA, int LDW, int LDC>
__global__ void gemm_bf16_wmma(const unsigned short* __restrict__ A,
                               const unsigned short* __restrict__ W,
                               const float* __restrict__ R,
                               float* __restrict__ C) {
    constexpr int groupsN = NN >> 6;
    constexpr int groupsM = MM >> 5;
    constexpr int totalWork = groupsM * groupsN;

    const int waveInBlock = threadIdx.x >> 5;            // 0..7
    const int work = blockIdx.x * 8 + waveInBlock;       // wave-uniform
    if (work >= totalWork) return;                       // whole wave exits

    const int gN = work % groupsN;
    const int gM = work / groupsN;

    const int lane = threadIdx.x & 31;
    const int half = lane >> 4;      // K-group select per ISA layout
    const int l16  = lane & 15;

    const unsigned short* __restrict__ aPtr =
        A + (size_t)(gM * 32 + l16) * LDA + half * 8;
    const unsigned short* __restrict__ wPtr =
        W + (size_t)(gN * 64 + l16) * LDW + half * 8;

    v8f acc00 = {0.f,0.f,0.f,0.f,0.f,0.f,0.f,0.f};
    v8f acc01 = acc00, acc02 = acc00, acc03 = acc00;
    v8f acc10 = acc00, acc11 = acc00, acc12 = acc00, acc13 = acc00;

    for (int kb = 0; kb < KK; kb += 32) {
        __builtin_prefetch(aPtr + 32, 0, 1);
        __builtin_prefetch(wPtr + 32, 0, 1);
        const Frag16 a0 = load_frag(aPtr);
        const Frag16 a1 = load_frag(aPtr + 16 * LDA);
        const Frag16 b0 = load_frag(wPtr);
        const Frag16 b1 = load_frag(wPtr + 16 * LDW);
        const Frag16 b2 = load_frag(wPtr + 32 * LDW);
        const Frag16 b3 = load_frag(wPtr + 48 * LDW);
        acc00 = __builtin_amdgcn_wmma_f32_16x16x32_bf16(false, a0.v, false, b0.v, (short)0, acc00, false, false);
        acc01 = __builtin_amdgcn_wmma_f32_16x16x32_bf16(false, a0.v, false, b1.v, (short)0, acc01, false, false);
        acc02 = __builtin_amdgcn_wmma_f32_16x16x32_bf16(false, a0.v, false, b2.v, (short)0, acc02, false, false);
        acc03 = __builtin_amdgcn_wmma_f32_16x16x32_bf16(false, a0.v, false, b3.v, (short)0, acc03, false, false);
        acc10 = __builtin_amdgcn_wmma_f32_16x16x32_bf16(false, a1.v, false, b0.v, (short)0, acc10, false, false);
        acc11 = __builtin_amdgcn_wmma_f32_16x16x32_bf16(false, a1.v, false, b1.v, (short)0, acc11, false, false);
        acc12 = __builtin_amdgcn_wmma_f32_16x16x32_bf16(false, a1.v, false, b2.v, (short)0, acc12, false, false);
        acc13 = __builtin_amdgcn_wmma_f32_16x16x32_bf16(false, a1.v, false, b3.v, (short)0, acc13, false, false);
        aPtr += 32;
        wPtr += 32;
    }

    // D layout: VGPR r, lanes 0-15 -> row r, lanes 16-31 -> row r+8
    const int colBase = gN * 64 + l16;
#pragma unroll
    for (int r = 0; r < 8; ++r) {
        const int mr = gM * 32 + r + half * 8;
        const size_t base = (size_t)mr * LDC + colBase;
        float v0 = acc00[r], v1 = acc01[r], v2 = acc02[r], v3 = acc03[r];
        if (HASR) {
            v0 += R[base]; v1 += R[base + 16]; v2 += R[base + 32]; v3 += R[base + 48];
        }
        C[base] = v0; C[base + 16] = v1; C[base + 32] = v2; C[base + 48] = v3;
    }
#pragma unroll
    for (int r = 0; r < 8; ++r) {
        const int mr = gM * 32 + 16 + r + half * 8;
        const size_t base = (size_t)mr * LDC + colBase;
        float v0 = acc10[r], v1 = acc11[r], v2 = acc12[r], v3 = acc13[r];
        if (HASR) {
            v0 += R[base]; v1 += R[base + 16]; v2 += R[base + 32]; v3 += R[base + 48];
        }
        C[base] = v0; C[base + 16] = v1; C[base + 32] = v2; C[base + 48] = v3;
    }
}

// ---------------------------------------------------------------------------
// f32 -> bf16 bulk converter (grid-stride)
// ---------------------------------------------------------------------------
__global__ void convert_bf16_kernel(const float* __restrict__ src,
                                    unsigned short* __restrict__ dst, int n) {
    for (int i = blockIdx.x * blockDim.x + threadIdx.x; i < n;
         i += gridDim.x * blockDim.x)
        dst[i] = f32_to_bf16(src[i]);
}

// ---------------------------------------------------------------------------
// Embedding + positional embedding
// ---------------------------------------------------------------------------
__global__ void embed_kernel(const int* __restrict__ ids,
                             const float* __restrict__ emb,
                             const float* __restrict__ pos,
                             float* __restrict__ h) {
    const int i = blockIdx.x * blockDim.x + threadIdx.x;
    if (i >= LSEQ * DMODEL) return;
    const int t = i / DMODEL;
    const int d = i - t * DMODEL;
    h[i] = emb[(size_t)ids[t] * DMODEL + d] + pos[i];
}

// ---------------------------------------------------------------------------
// RMSNorm -> bf16 output (only feeds GEMMs). One block per row.
// ---------------------------------------------------------------------------
__global__ void rmsnorm_bf16_kernel(const float* __restrict__ x,
                                    const float* __restrict__ w,
                                    unsigned short* __restrict__ out, int D) {
    const int row = blockIdx.x;
    const int tid = threadIdx.x;
    const float* __restrict__ xr = x + (size_t)row * D;
    float s = 0.f;
    for (int d = tid; d < D; d += 256) { float v = xr[d]; s += v * v; }
#pragma unroll
    for (int off = 16; off >= 1; off >>= 1) s += __shfl_xor(s, off, 32);
    __shared__ float red[8];
    __shared__ float invs;
    if ((tid & 31) == 0) red[tid >> 5] = s;
    __syncthreads();
    if (tid == 0) {
        float tot = 0.f;
#pragma unroll
        for (int i = 0; i < 8; ++i) tot += red[i];
        invs = rsqrtf(tot / (float)D + 1e-6f);
    }
    __syncthreads();
    const float inv = invs;
    unsigned short* __restrict__ orow = out + (size_t)row * D;
    for (int d = tid; d < D; d += 256) orow[d] = f32_to_bf16(xr[d] * inv * w[d]);
}

// ---------------------------------------------------------------------------
// Depthwise causal conv (width 4) + bias + SiLU; writes f32 + bf16
// ---------------------------------------------------------------------------
__global__ void conv_silu_kernel(const float* __restrict__ xz,
                                 const float* __restrict__ cw,
                                 const float* __restrict__ cb,
                                 float* __restrict__ u,
                                 unsigned short* __restrict__ ubf) {
    const int i = blockIdx.x * blockDim.x + threadIdx.x;
    if (i >= LSEQ * DINNER) return;
    const int t = i / DINNER;
    const int d = i - t * DINNER;
    float s = cb[d];
#pragma unroll
    for (int j = 0; j < DCONV; ++j) {
        const int tt = t - (DCONV - 1) + j;
        if (tt >= 0) s += xz[(size_t)tt * DXZ + d] * cw[d * DCONV + j];
    }
    const float r = s * sigmoidf_(s);
    u[i]   = r;
    ubf[i] = f32_to_bf16(r);
}

// ---------------------------------------------------------------------------
// Extract bf16 dt = xdbl[:, 0:DTRANK] for the delta GEMM
// ---------------------------------------------------------------------------
__global__ void dt_extract_kernel(const float* __restrict__ xdbl,
                                  unsigned short* __restrict__ dtbf) {
    const int i = blockIdx.x * blockDim.x + threadIdx.x;
    if (i >= LSEQ * DTRANK) return;
    const int t = i / DTRANK;
    const int r = i - t * DTRANK;
    dtbf[i] = f32_to_bf16(xdbl[(size_t)t * NXDBL + r]);
}

// ---------------------------------------------------------------------------
// delta = softplus(raw + dt_b[d])
// ---------------------------------------------------------------------------
__global__ void softplus_bias_kernel(float* __restrict__ delta,
                                     const float* __restrict__ db) {
    const int i = blockIdx.x * blockDim.x + threadIdx.x;
    if (i >= LSEQ * DINNER) return;
    const int d = i % DINNER;
    const float x = delta[i] + db[d];
    delta[i] = (x > 20.f) ? x : log1pf(__expf(x));
}

// ---------------------------------------------------------------------------
// Selective scan. One lane per (channel, state); 16-lane shfl_xor reduction;
// epilogue fuses (y + u*Dp) * silu(z), writes bf16 y.
// ---------------------------------------------------------------------------
__global__ void scan_kernel(const float* __restrict__ delta,
                            const float* __restrict__ u,
                            const float* __restrict__ xdbl,   // [L, 64]
                            const float* __restrict__ xz,     // z at cols 1024+
                            const float* __restrict__ A_log,  // [DINNER, 16]
                            const float* __restrict__ Dp,     // [DINNER]
                            unsigned short* __restrict__ yout) {
    const int tid = threadIdx.x;
    const int s = tid & 15;
    const int cLocal = tid >> 4;
    const int d = blockIdx.x * 16 + cLocal;

    const float Ads = -__expf(A_log[(size_t)d * DSTATE + s]);
    const float Dpd = Dp[d];
    float hs = 0.f;

    for (int t = 0; t < LSEQ; ++t) {
        const float dlt = delta[(size_t)t * DINNER + d];
        const float uu  = u[(size_t)t * DINNER + d];
        const float Bv  = xdbl[(size_t)t * NXDBL + DTRANK + s];
        const float Cv  = xdbl[(size_t)t * NXDBL + DTRANK + DSTATE + s];
        hs = __expf(dlt * Ads) * hs + dlt * Bv * uu;
        float y = hs * Cv;
#pragma unroll
        for (int off = 1; off <= 8; off <<= 1) y += __shfl_xor(y, off, 32);
        if (s == 0) {
            const float zz = xz[(size_t)t * DXZ + DINNER + d];
            yout[(size_t)t * DINNER + d] =
                f32_to_bf16((y + uu * Dpd) * (zz * sigmoidf_(zz)));
        }
    }
}

// ---------------------------------------------------------------------------
// Per-row NLL
// ---------------------------------------------------------------------------
__global__ void row_nll_kernel(const float* __restrict__ logits,
                               const int* __restrict__ labels,
                               float* __restrict__ nll) {
    const int t = blockIdx.x;            // 0 .. L-2
    const int tid = threadIdx.x;
    const float* __restrict__ row = logits + (size_t)t * VOCAB;
    const int target = labels[t + 1];

    __shared__ float red[8];
    __shared__ float bmax, bsum;

    float mx = -3.402823466e38f;
    for (int v = tid; v < VOCAB; v += 256) mx = fmaxf(mx, row[v]);
#pragma unroll
    for (int off = 16; off >= 1; off >>= 1) mx = fmaxf(mx, __shfl_xor(mx, off, 32));
    if ((tid & 31) == 0) red[tid >> 5] = mx;
    __syncthreads();
    if (tid == 0) {
        float m = red[0];
#pragma unroll
        for (int i = 1; i < 8; ++i) m = fmaxf(m, red[i]);
        bmax = m;
    }
    __syncthreads();
    const float m = bmax;

    float se = 0.f;
    for (int v = tid; v < VOCAB; v += 256) se += __expf(row[v] - m);
#pragma unroll
    for (int off = 16; off >= 1; off >>= 1) se += __shfl_xor(se, off, 32);
    if ((tid & 31) == 0) red[tid >> 5] = se;
    __syncthreads();
    if (tid == 0) {
        float tot = 0.f;
#pragma unroll
        for (int i = 0; i < 8; ++i) tot += red[i];
        bsum = tot;
    }
    __syncthreads();
    if (tid == 0) {
        const bool valid = (target != -100);
        nll[t] = valid ? ((__logf(bsum) + m) - row[target]) : 0.f;
    }
}

// ---------------------------------------------------------------------------
// Final loss reduction
// ---------------------------------------------------------------------------
__global__ void loss_reduce_kernel(const float* __restrict__ nll,
                                   const int* __restrict__ labels,
                                   float* __restrict__ lossOut) {
    const int tid = threadIdx.x;
    float s = 0.f;
    float cnt = 0.f;
    for (int t = tid; t < LSEQ - 1; t += 256) {
        if (labels[t + 1] != -100) { s += nll[t]; cnt += 1.f; }
    }
#pragma unroll
    for (int off = 16; off >= 1; off >>= 1) {
        s += __shfl_xor(s, off, 32);
        cnt += __shfl_xor(cnt, off, 32);
    }
    __shared__ float reds[8], redc[8];
    if ((tid & 31) == 0) { reds[tid >> 5] = s; redc[tid >> 5] = cnt; }
    __syncthreads();
    if (tid == 0) {
        float ts = 0.f, tc = 0.f;
#pragma unroll
        for (int i = 0; i < 8; ++i) { ts += reds[i]; tc += redc[i]; }
        lossOut[0] = ts / fmaxf(tc, 1.f);
    }
}

// ---------------------------------------------------------------------------
// Host-side orchestration
// ---------------------------------------------------------------------------
extern "C" void kernel_launch(void* const* d_in, const int* in_sizes, int n_in,
                              void* d_out, int out_size, void* d_ws, size_t ws_size,
                              hipStream_t stream) {
    (void)in_sizes; (void)n_in; (void)out_size; (void)ws_size;

    const int*   input_ids    = (const int*)  d_in[0];
    const int*   labels       = (const int*)  d_in[1];
    const float* emb          = (const float*)d_in[2];
    const float* pos_emb      = (const float*)d_in[3];
    const float* final_norm_w = (const float*)d_in[4];
    const float* norm_w       = (const float*)d_in[5];
    const float* in_w         = (const float*)d_in[6];
    const float* conv_w       = (const float*)d_in[7];
    const float* conv_b       = (const float*)d_in[8];
    const float* xproj_w      = (const float*)d_in[9];
    const float* dt_w         = (const float*)d_in[10];
    const float* dt_b         = (const float*)d_in[11];
    const float* A_log        = (const float*)d_in[12];
    const float* Dp           = (const float*)d_in[13];
    const float* out_w        = (const float*)d_in[14];

    float* logits = (float*)d_out;                         // [L, VOCAB]
    float* lossP  = logits + (size_t)LSEQ * VOCAB;         // [1]

    // ----- workspace carve-up -----
    float* ws    = (float*)d_ws;
    float* h     = ws;                              // [L, DMODEL]
    float* xz    = h     + (size_t)LSEQ * DMODEL;   // [L, DXZ]
    float* u     = xz    + (size_t)LSEQ * DXZ;      // [L, DINNER]
    float* xdbl  = u     + (size_t)LSEQ * DINNER;   // [L, NXDBL]
    float* delta = xdbl  + (size_t)LSEQ * NXDBL;    // [L, DINNER]
    float* nll   = delta + (size_t)LSEQ * DINNER;   // [L]
    unsigned short* bws     = (unsigned short*)(nll + 2048);
    unsigned short* in_w_bf = bws;                                           // 8*DXZ*DMODEL
    unsigned short* xw_bf   = in_w_bf + (size_t)NLAYERS * DXZ * DMODEL;      // 8*NXDBL*DINNER
    unsigned short* dtw_bf  = xw_bf   + (size_t)NLAYERS * NXDBL * DINNER;    // 8*DINNER*DTRANK
    unsigned short* ow_bf   = dtw_bf  + (size_t)NLAYERS * DINNER * DTRANK;   // 8*DMODEL*DINNER
    unsigned short* emb_bf  = ow_bf   + (size_t)NLAYERS * DMODEL * DINNER;   // VOCAB*DMODEL
    unsigned short* xbuf_bf = emb_bf  + (size_t)VOCAB * DMODEL;              // L*DMODEL
    unsigned short* u_bf    = xbuf_bf + (size_t)LSEQ * DMODEL;               // L*DINNER
    unsigned short* dt_bf   = u_bf    + (size_t)LSEQ * DINNER;               // L*DTRANK
    unsigned short* y_bf    = dt_bf   + (size_t)LSEQ * DTRANK;               // L*DINNER

    const dim3 blk256(256);
    auto cblocks = [](size_t n) { return (unsigned)((n + 255) / 256); };
    auto gblocks = [](int M, int N) { return ((M >> 5) * (N >> 6) + 7) / 8; };

    // ----- weight conversion to bf16 -----
    convert_bf16_kernel<<<4096, blk256, 0, stream>>>(in_w,    in_w_bf, NLAYERS * DXZ * DMODEL);
    convert_bf16_kernel<<<1024, blk256, 0, stream>>>(xproj_w, xw_bf,   NLAYERS * NXDBL * DINNER);
    convert_bf16_kernel<<<512,  blk256, 0, stream>>>(dt_w,    dtw_bf,  NLAYERS * DINNER * DTRANK);
    convert_bf16_kernel<<<4096, blk256, 0, stream>>>(out_w,   ow_bf,   NLAYERS * DMODEL * DINNER);
    convert_bf16_kernel<<<2048, blk256, 0, stream>>>(emb,     emb_bf,  VOCAB * DMODEL);

    // embedding
    embed_kernel<<<cblocks((size_t)LSEQ * DMODEL), blk256, 0, stream>>>(
        input_ids, emb, pos_emb, h);

    for (int i = 0; i < NLAYERS; ++i) {
        const float*          nw = norm_w  + (size_t)i * DMODEL;
        const unsigned short* iw = in_w_bf + (size_t)i * DXZ * DMODEL;
        const float*          cw = conv_w  + (size_t)i * DINNER * DCONV;
        const float*          cb = conv_b  + (size_t)i * DINNER;
        const unsigned short* xw = xw_bf   + (size_t)i * NXDBL * DINNER;
        const unsigned short* dw = dtw_bf  + (size_t)i * DINNER * DTRANK;
        const float*          db = dt_b    + (size_t)i * DINNER;
        const float*          al = A_log   + (size_t)i * DINNER * DSTATE;
        const float*          dp = Dp      + (size_t)i * DINNER;
        const unsigned short* ow = ow_bf   + (size_t)i * DMODEL * DINNER;

        rmsnorm_bf16_kernel<<<LSEQ, blk256, 0, stream>>>(h, nw, xbuf_bf, DMODEL);

        // xz = x @ in_w^T  [2048 x 2048 x 512]
        gemm_bf16_wmma<false, LSEQ, DXZ, DMODEL, DMODEL, DMODEL, DXZ>
            <<<gblocks(LSEQ, DXZ), blk256, 0, stream>>>(xbuf_bf, iw, nullptr, xz);

        conv_silu_kernel<<<cblocks((size_t)LSEQ * DINNER), blk256, 0, stream>>>(
            xz, cw, cb, u, u_bf);

        // xdbl = u @ xproj_w^T  [2048 x 64 x 1024]
        gemm_bf16_wmma<false, LSEQ, NXDBL, DINNER, DINNER, DINNER, NXDBL>
            <<<gblocks(LSEQ, NXDBL), blk256, 0, stream>>>(u_bf, xw, nullptr, xdbl);

        dt_extract_kernel<<<cblocks((size_t)LSEQ * DTRANK), blk256, 0, stream>>>(
            xdbl, dt_bf);

        // delta_raw = dt @ dt_w^T  [2048 x 1024 x 32]
        gemm_bf16_wmma<false, LSEQ, DINNER, DTRANK, DTRANK, DTRANK, DINNER>
            <<<gblocks(LSEQ, DINNER), blk256, 0, stream>>>(dt_bf, dw, nullptr, delta);

        softplus_bias_kernel<<<cblocks((size_t)LSEQ * DINNER), blk256, 0, stream>>>(
            delta, db);

        scan_kernel<<<DINNER / 16, blk256, 0, stream>>>(
            delta, u, xdbl, xz, al, dp, y_bf);

        // h = h + y @ out_w^T  [2048 x 512 x 1024]
        gemm_bf16_wmma<true, LSEQ, DMODEL, DINNER, DINNER, DINNER, DMODEL>
            <<<gblocks(LSEQ, DMODEL), blk256, 0, stream>>>(y_bf, ow, h, h);
    }

    rmsnorm_bf16_kernel<<<LSEQ, blk256, 0, stream>>>(h, final_norm_w, xbuf_bf, DMODEL);

    // logits = x_final @ emb^T  [2048 x 4096 x 512] -> d_out
    gemm_bf16_wmma<false, LSEQ, VOCAB, DMODEL, DMODEL, DMODEL, VOCAB>
        <<<gblocks(LSEQ, VOCAB), blk256, 0, stream>>>(xbuf_bf, emb_bf, nullptr, logits);

    // loss
    row_nll_kernel<<<LSEQ - 1, blk256, 0, stream>>>(logits, labels, nll);
    loss_reduce_kernel<<<1, blk256, 0, stream>>>(nll, labels, lossP);
}
